// Head_34445637714363
// MI455X (gfx1250) — matrix operand order repaired and checked by reading, not compiled, not run on hardware
//
#include <hip/hip_runtime.h>
#include <hip/hip_bf16.h>

#define EMBEDC 1024
#define HEADD  128
#define TSEQ   2048
#define BATCHN 8
#define BT     (BATCHN * TSEQ)   // 16384

typedef __bf16 bf16_t;
typedef __attribute__((ext_vector_type(16))) __bf16 bf16x16;
typedef __attribute__((ext_vector_type(8)))  __bf16 bf16x8;
typedef __attribute__((ext_vector_type(8)))  float  f32x8;
typedef __attribute__((ext_vector_type(4)))  float  f32x4;

union Frag16 { bf16x16 v; bf16x8 h[2]; };

__device__ __forceinline__ f32x8 wmma_bf16(bf16x16 a, bf16x16 b, f32x8 c) {
    // v_wmma_f32_16x16x32_bf16: D = A(16x32) * B(32x16) + C(16x16 f32)
    return __builtin_amdgcn_wmma_f32_16x16x32_bf16(
        /*neg_a=*/false, a, /*neg_b=*/false, b,
        /*c_mod=*/(short)0, c, /*reuse_a=*/false, /*reuse_b=*/false);
}

// Load 8 consecutive f32 and convert to bf16x8 (two b128 loads + 4 pack cvts)
__device__ __forceinline__ bf16x8 cvt8(const float* __restrict__ p) {
    f32x4 a = *(const f32x4*)p;
    f32x4 b = *(const f32x4*)(p + 4);
    bf16x8 r;
    r[0] = (__bf16)a[0]; r[1] = (__bf16)a[1]; r[2] = (__bf16)a[2]; r[3] = (__bf16)a[3];
    r[4] = (__bf16)b[0]; r[5] = (__bf16)b[1]; r[6] = (__bf16)b[2]; r[7] = (__bf16)b[3];
    return r;
}

// ---------------------------------------------------------------------------
// Kernel 1a: convert x (f32) -> xb (bf16), one pass. 8 elements per thread.
// ---------------------------------------------------------------------------
__global__ __launch_bounds__(256) void prep_x(
    const float* __restrict__ x, bf16_t* __restrict__ xb)
{
    long i = ((long)blockIdx.x * blockDim.x + threadIdx.x) * 8;
    if (i >= (long)BT * EMBEDC) return;
    *(bf16x8*)(xb + i) = cvt8(x + i);
}

// ---------------------------------------------------------------------------
// Kernel 1b: transpose + convert W[E][H] -> WT[H][E] (bf16) for q,k,v so the
// projection B-fragments are contiguous per lane.
// ---------------------------------------------------------------------------
__global__ __launch_bounds__(256) void prep_wt(
    const float* __restrict__ Wq, const float* __restrict__ Wk,
    const float* __restrict__ Wv,
    bf16_t* __restrict__ WTq, bf16_t* __restrict__ WTk, bf16_t* __restrict__ WTv)
{
    const int total = HEADD * EMBEDC;               // 131072
    int idx = blockIdx.x * blockDim.x + threadIdx.x;
    if (idx >= 3 * total) return;
    int m = idx / total;
    int r = idx - m * total;
    int h = r / EMBEDC;
    int e = r - h * EMBEDC;
    const float* W = (m == 0) ? Wq : (m == 1) ? Wk : Wv;
    bf16_t*      D = (m == 0) ? WTq : (m == 1) ? WTk : WTv;
    D[r] = (bf16_t)W[(long)e * HEADD + h];
}

// ---------------------------------------------------------------------------
// Kernel 2: projections. One wave computes a full 16(tokens) x 128(headdim)
// strip of one matrix. Per K-step: A-fragment + all 8 B-fragments are loaded
// into independent registers first (one clause train, one wait), then the 8
// WMMAs issue back-to-back.
// q,k stored row-major bf16 [BT][H]; v stored transposed bf16 [B][H][T].
// ---------------------------------------------------------------------------
__global__ __launch_bounds__(256) void proj_kernel(
    const bf16_t* __restrict__ xb,
    const bf16_t* __restrict__ WTq, const bf16_t* __restrict__ WTk,
    const bf16_t* __restrict__ WTv,
    bf16_t* __restrict__ qo, bf16_t* __restrict__ ko, bf16_t* __restrict__ vT)
{
    const int lane = threadIdx.x & 31;
    const int wid  = (blockIdx.x * blockDim.x + threadIdx.x) >> 5;
    const int mtile = wid / 3;           // 1024 token tiles
    const int mat   = wid - mtile * 3;   // 0=q, 1=k, 2=v
    const bf16_t* WT = (mat == 0) ? WTq : (mat == 1) ? WTk : WTv;

    const int g = lane >> 4;             // lane half
    const int n = lane & 15;
    const long tokBase = (long)mtile * 16;
    const int kofsA = g ? 16 : 0;        // upper lanes hold K 16..31

    // A source: xb row for this lane (A rows = lane%16 in both halves)
    const bf16_t* xrow = xb + (tokBase + n) * EMBEDC;
    // B source: WT row (col = lane%16), 16 consecutive K per lane half
    const bf16_t* wcol = WT + (long)n * EMBEDC + (g ? 16 : 0);

    f32x8 acc[8];
    #pragma unroll
    for (int nt = 0; nt < 8; ++nt)
        acc[nt] = (f32x8){0.f,0.f,0.f,0.f,0.f,0.f,0.f,0.f};

    for (int e = 0; e < EMBEDC; e += 32) {
        Frag16 a;
        Frag16 bfr[8];
        a.h[0] = *(const bf16x8*)(xrow + e + kofsA);
        a.h[1] = *(const bf16x8*)(xrow + e + 8 + kofsA);
        #pragma unroll
        for (int nt = 0; nt < 8; ++nt) {
            const bf16_t* wr = wcol + (long)nt * 16 * EMBEDC + e;
            bfr[nt].h[0] = *(const bf16x8*)(wr);
            bfr[nt].h[1] = *(const bf16x8*)(wr + 8);
        }
        #pragma unroll
        for (int nt = 0; nt < 8; ++nt)
            acc[nt] = wmma_bf16(a.v, bfr[nt].v, acc[nt]);
    }

    if (mat < 2) {
        bf16_t* out = (mat == 0) ? qo : ko;  // row-major [BT][H]
        #pragma unroll
        for (int nt = 0; nt < 8; ++nt)
            #pragma unroll
            for (int r = 0; r < 8; ++r) {
                long tok = tokBase + r + 8 * g;   // C layout: row = vgpr + 8*half
                out[tok * HEADD + nt * 16 + n] = (bf16_t)acc[nt][r];
            }
    } else {
        // vT[b][h][t]; lane holds column h = nt*16+n, 8 consecutive t
        int b    = (int)(tokBase / TSEQ);
        int tloc = (int)(tokBase - (long)b * TSEQ) + 8 * g;
        #pragma unroll
        for (int nt = 0; nt < 8; ++nt) {
            bf16x8 pack;
            #pragma unroll
            for (int r = 0; r < 8; ++r) pack[r] = (bf16_t)acc[nt][r];
            *(bf16x8*)(vT + ((long)b * HEADD + nt * 16 + n) * TSEQ + tloc) = pack;
        }
    }
}

// ---------------------------------------------------------------------------
// Kernel 3: flash attention. One wave per (batch, 16-query-row tile).
// K/V fragments are batched into independent registers before each WMMA
// burst so the loads pipeline instead of serializing per-WMMA.
// ---------------------------------------------------------------------------
__global__ __launch_bounds__(128) void attn_kernel(
    const bf16_t* __restrict__ q, const bf16_t* __restrict__ k,
    const bf16_t* __restrict__ vT, float* __restrict__ out)
{
    __shared__ __align__(16) __bf16 pbuf[4][16 * 32];  // per-wave P tile 16x32

    const int lane = threadIdx.x & 31;
    const int w    = threadIdx.x >> 5;
    const int wid  = blockIdx.x * 4 + w;
    const int b    = wid >> 7;          // / (T/16 = 128)
    const int qt   = wid & 127;
    const int g    = lane >> 4;
    const int n    = lane & 15;
    const int kofsA = g ? 16 : 0;

    // Load Q tile as four 16x32 A-fragments (kept in registers)
    const bf16_t* qrow = q + ((long)b * TSEQ + qt * 16 + n) * HEADD;
    Frag16 Qa[4];
    #pragma unroll
    for (int kk = 0; kk < 4; ++kk) {
        Qa[kk].h[0] = *(const bf16x8*)(qrow + kk * 32 + kofsA);
        Qa[kk].h[1] = *(const bf16x8*)(qrow + kk * 32 + 8 + kofsA);
    }

    f32x8 acc[8];
    #pragma unroll
    for (int h = 0; h < 8; ++h) acc[h] = (f32x8){0.f,0.f,0.f,0.f,0.f,0.f,0.f,0.f};
    float mrow[8], lrow[8];
    #pragma unroll
    for (int r = 0; r < 8; ++r) { mrow[r] = -1.0e30f; lrow[r] = 0.f; }

    const bf16_t* kb = k  + (long)b * TSEQ * HEADD;
    const bf16_t* vb = vT + (long)b * HEADD * TSEQ;
    const int nPairs = (qt >> 1) + 1;   // 32-wide key steps up to causal limit

    for (int jj = 0; jj < nPairs; ++jj) {
        const int s0 = jj * 32;
        f32x8 cs[2];
        #pragma unroll
        for (int st = 0; st < 2; ++st) {
            const int sb = s0 + st * 16;
            const bf16_t* krow = kb + (long)(sb + n) * HEADD + (g ? 16 : 0);
            // batch the 4 K-fragments, then 4 back-to-back WMMAs
            Frag16 kf[4];
            #pragma unroll
            for (int kk = 0; kk < 4; ++kk) {
                kf[kk].h[0] = *(const bf16x8*)(krow + kk * 32);
                kf[kk].h[1] = *(const bf16x8*)(krow + kk * 32 + 8);
            }
            f32x8 c = {0.f,0.f,0.f,0.f,0.f,0.f,0.f,0.f};
            #pragma unroll
            for (int kk = 0; kk < 4; ++kk)
                c = wmma_bf16(Qa[kk].v, kf[kk].v, c);
            // scale by 1/sqrt(EMBED) and apply causal mask
            #pragma unroll
            for (int r = 0; r < 8; ++r) {
                float sc   = c[r] * 0.03125f;
                int   trow = qt * 16 + r + 8 * g;
                int   scol = sb + n;
                cs[st][r] = (scol > trow) ? -1.0e30f : sc;
            }
        }
        // online softmax: row max across the 32-wide pair (reduce over N lanes)
        float alpha[8];
        #pragma unroll
        for (int r = 0; r < 8; ++r) {
            float vmax = fmaxf(cs[0][r], cs[1][r]);
            #pragma unroll
            for (int msk = 1; msk < 16; msk <<= 1)
                vmax = fmaxf(vmax, __shfl_xor(vmax, msk, 32));
            float mn = fmaxf(mrow[r], vmax);
            alpha[r] = __expf(mrow[r] - mn);
            mrow[r]  = mn;
        }
        // exponentiate + row sums
        #pragma unroll
        for (int r = 0; r < 8; ++r) {
            float p0 = __expf(cs[0][r] - mrow[r]);
            float p1 = __expf(cs[1][r] - mrow[r]);
            cs[0][r] = p0; cs[1][r] = p1;
            float s = p0 + p1;
            #pragma unroll
            for (int msk = 1; msk < 16; msk <<= 1)
                s += __shfl_xor(s, msk, 32);
            lrow[r] = lrow[r] * alpha[r] + s;
        }
        // rescale running accumulator
        #pragma unroll
        for (int h = 0; h < 8; ++h)
            #pragma unroll
            for (int r = 0; r < 8; ++r) acc[h][r] *= alpha[r];

        // C-layout -> A-layout for P via per-wave LDS bounce (16x32 bf16)
        __bf16* pb = pbuf[w];
        #pragma unroll
        for (int st = 0; st < 2; ++st)
            #pragma unroll
            for (int r = 0; r < 8; ++r)
                pb[(r + 8 * g) * 32 + st * 16 + n] = (__bf16)cs[st][r];
        asm volatile("s_wait_dscnt 0" ::: "memory");
        Frag16 Pa;
        Pa.h[0] = *(const bf16x8*)(pb + n * 32 + kofsA);
        Pa.h[1] = *(const bf16x8*)(pb + n * 32 + 8 + kofsA);

        // acc += P(16x32) @ V(32x16); V fragments batched 4 at a time
        #pragma unroll
        for (int hb = 0; hb < 2; ++hb) {
            Frag16 Vf[4];
            #pragma unroll
            for (int hh = 0; hh < 4; ++hh) {
                const int h = hb * 4 + hh;
                const bf16_t* vr = vb + (long)(h * 16 + n) * TSEQ + s0 + (g ? 16 : 0);
                Vf[hh].h[0] = *(const bf16x8*)(vr);
                Vf[hh].h[1] = *(const bf16x8*)(vr + 8);
            }
            #pragma unroll
            for (int hh = 0; hh < 4; ++hh)
                acc[hb * 4 + hh] = wmma_bf16(Pa.v, Vf[hh].v, acc[hb * 4 + hh]);
        }
    }

    // epilogue: out[b][t][h] = acc / l   (f32)
    float* orow = out + ((long)b * TSEQ + qt * 16) * HEADD;
    #pragma unroll
    for (int h = 0; h < 8; ++h) {
        #pragma unroll
        for (int r = 0; r < 8; ++r) {
            float inv = 1.0f / lrow[r];
            orow[(long)(r + 8 * g) * HEADD + h * 16 + n] = acc[h][r] * inv;
        }
    }
}

// ---------------------------------------------------------------------------
extern "C" void kernel_launch(void* const* d_in, const int* in_sizes, int n_in,
                              void* d_out, int out_size, void* d_ws, size_t ws_size,
                              hipStream_t stream) {
    // setup_inputs order: x, Wk, Wq, Wv
    const float* x  = (const float*)d_in[0];
    const float* Wk = (const float*)d_in[1];
    const float* Wq = (const float*)d_in[2];
    const float* Wv = (const float*)d_in[3];
    float* out = (float*)d_out;

    char* ws = (char*)d_ws;
    const size_t XB_BYTES = (size_t)BT * EMBEDC * sizeof(bf16_t);    // 32 MB
    const size_t WT_BYTES = (size_t)HEADD * EMBEDC * sizeof(bf16_t); // 256 KB
    const size_t P_BYTES  = (size_t)BT * HEADD * sizeof(bf16_t);     // 4 MB
    bf16_t* xb  = (bf16_t*)(ws);
    bf16_t* WTq = (bf16_t*)(ws + XB_BYTES);
    bf16_t* WTk = (bf16_t*)(ws + XB_BYTES + WT_BYTES);
    bf16_t* WTv = (bf16_t*)(ws + XB_BYTES + 2 * WT_BYTES);
    bf16_t* qb  = (bf16_t*)(ws + XB_BYTES + 3 * WT_BYTES);
    bf16_t* kbf = (bf16_t*)(ws + XB_BYTES + 3 * WT_BYTES + P_BYTES);
    bf16_t* vT  = (bf16_t*)(ws + XB_BYTES + 3 * WT_BYTES + 2 * P_BYTES);

    // 1a) x f32 -> bf16 (single pass)
    {
        long n8 = (long)BT * EMBEDC / 8;          // 2,097,152 threads
        prep_x<<<(int)((n8 + 255) / 256), 256, 0, stream>>>(x, xb);
    }
    // 1b) transpose+convert weights
    {
        int total = 3 * HEADD * EMBEDC;
        prep_wt<<<(total + 255) / 256, 256, 0, stream>>>(Wq, Wk, Wv, WTq, WTk, WTv);
    }
    // 2) projections: one wave per (token-tile, matrix): 1024*3 waves, 8/block
    {
        int waves  = (BT / 16) * 3;      // 3072
        int blocks = waves / 8;          // 384
        proj_kernel<<<blocks, 256, 0, stream>>>(xb, WTq, WTk, WTv, qb, kbf, vT);
    }
    // 3) flash attention: B*(T/16) = 1024 waves, 4 waves/block
    {
        attn_kernel<<<256, 128, 0, stream>>>(qb, kbf, vT, out);
    }
}
